// SwinTransformerBlock_44521630991198
// MI455X (gfx1250) — matrix operand order repaired and checked by reading, not compiled, not run on hardware
//
#include <hip/hip_runtime.h>
#include <hip/hip_bf16.h>

typedef __attribute__((ext_vector_type(16))) _Float16 v16h;
typedef __attribute__((ext_vector_type(8)))  _Float16 v8h;
typedef __attribute__((ext_vector_type(8)))  float    v8f;
typedef __attribute__((ext_vector_type(4)))  unsigned v4u;
typedef __attribute__((ext_vector_type(8)))  unsigned v8u;

#define DEVI __device__ __forceinline__

constexpr int Bn   = 64;
constexpr int Hh   = 56;
constexpr int Wd   = 56;
constexpr int Cc   = 96;
constexpr int SH3  = 3;
constexpr int NH   = 3;
constexpr int Ntok = 49;            // tokens per window
constexpr int HD   = 32;            // head dim
constexpr int MLPH = 384;
constexpr int NWTOT = Bn * 64;      // 4096 windows total
constexpr int ROWS  = Bn * Hh * Wd; // 200704 token rows

// ---------------- WMMA fragment loaders (CDNA5 16-bit layouts) ----------------
// A 16x32 f16: lane L<16 -> row L holds K 0..7 & 16..23; lane L>=16 -> row L-16,
// K 8..15 & 24..31.
DEVI v16h frag_a(const _Float16* As, int lda, int row0, int k0) {
  int lane = threadIdx.x & 31;
  const _Float16* p = As + (size_t)(row0 + (lane & 15)) * lda + k0 + ((lane >> 4) << 3);
  v8h x0 = *(const v8h*)(p);
  v8h x1 = *(const v8h*)(p + 16);
  v16h a;
#pragma unroll
  for (int i = 0; i < 8; ++i) { a[i] = x0[i]; a[i + 8] = x1[i]; }
  return a;
}

// B 32x16 f16 from an LDS-transposed copy BT[n][k]: same K split as frag_a.
DEVI v16h frag_bt(const _Float16* BTs, int ldb, int k0, int n0) {
  int lane = threadIdx.x & 31;
  const _Float16* p = BTs + (size_t)(n0 + (lane & 15)) * ldb + k0 + ((lane >> 4) << 3);
  v8h x0 = *(const v8h*)(p);
  v8h x1 = *(const v8h*)(p + 16);
  v16h b;
#pragma unroll
  for (int i = 0; i < 8; ++i) { b[i] = x0[i]; b[i + 8] = x1[i]; }
  return b;
}

DEVI v8f wmma16(v16h a, v16h b, v8f c) {
  return __builtin_amdgcn_wmma_f32_16x16x32_f16(false, a, false, b, (short)0, c,
                                                false, false);
}

DEVI float wave_sum(float v) {
#pragma unroll
  for (int m = 16; m >= 1; m >>= 1) v += __shfl_xor(v, m, 32);
  return v;
}
DEVI float wave_max(float v) {
#pragma unroll
  for (int m = 16; m >= 1; m >>= 1) v = fmaxf(v, __shfl_xor(v, m, 32));
  return v;
}

// ---------------- Tensor Data Mover: 2D f16 tile, global -> LDS ---------------
// D# per CDNA5 ISA. data_size=2B; tensor dims set
// equal to tile dims (exact fit, no OOB); strides in elements; addresses bytes.
DEVI void tdm_load_2d_f16(unsigned lds_addr, const void* gptr,
                          unsigned tile_w, unsigned tile_h, unsigned stride_el) {
  unsigned long long ga = (unsigned long long)(size_t)gptr;
  v4u g0;
  g0[0] = 1u;                                       // count=1, user mode
  g0[1] = lds_addr;                                 // LDS byte address
  g0[2] = (unsigned)ga;                             // global addr [31:0]
  g0[3] = (unsigned)(ga >> 32) | (2u << 30);        // addr [56:32] | type=2
  v8u g1;
  g1[0] = 1u << 16;                                 // data_size=1 (2 bytes)
  g1[1] = (tile_w & 0xffffu) << 16;                 // tensor_dim0[15:0]
  g1[2] = (tile_w >> 16) | ((tile_h & 0xffffu) << 16); // dim0 hi | tensor_dim1 lo
  g1[3] = (tile_h >> 16) | ((tile_w & 0xffffu) << 16); // dim1 hi | tile_dim0
  g1[4] = tile_h & 0xffffu;                         // tile_dim1 (tile_dim2=0)
  g1[5] = stride_el;                                // tensor_dim0_stride lo32
  g1[6] = 0u;
  g1[7] = 0u;
  asm volatile("tensor_load_to_lds %0, %1" :: "s"(g0), "s"(g1) : "memory");
}

// ---------------- Kernel 1: LN1 + shift + window + QKV GEMM ----------------
__global__ __launch_bounds__(256) void k_ln1_qkv(
    const float* __restrict__ x, const float* __restrict__ g1,
    const float* __restrict__ b1, const float* __restrict__ qkv_w,
    const float* __restrict__ qkv_b, _Float16* __restrict__ qkvbuf) {
  __shared__ __align__(32) _Float16 Xs[64][96];
  __shared__ __align__(32) _Float16 WT[144][96];
  int w = blockIdx.x;
  int b = w >> 6, widx = w & 63;
  int wh = widx >> 3, ww2 = widx & 7;
  int tid = threadIdx.x, wave = tid >> 5, lane = tid & 31;

#pragma unroll
  for (int i = 0; i < 8; ++i) {
    int r = wave * 8 + i;
    if (r < Ntok) {
      int th = r / 7, tw = r % 7;
      int sh = (wh * 7 + th + SH3) % Hh;
      int sw = (ww2 * 7 + tw + SH3) % Wd;
      const float* px = x + ((size_t)b * (Hh * Wd) + sh * Wd + sw) * Cc;
      float v0 = px[lane], v1 = px[lane + 32], v2 = px[lane + 64];
      float mu = wave_sum(v0 + v1 + v2) * (1.f / 96.f);
      float d0 = v0 - mu, d1 = v1 - mu, d2 = v2 - mu;
      float var = wave_sum(d0 * d0 + d1 * d1 + d2 * d2) * (1.f / 96.f);
      float rs = rsqrtf(var + 1e-5f);
      Xs[r][lane]      = (_Float16)(d0 * rs * g1[lane]      + b1[lane]);
      Xs[r][lane + 32] = (_Float16)(d1 * rs * g1[lane + 32] + b1[lane + 32]);
      Xs[r][lane + 64] = (_Float16)(d2 * rs * g1[lane + 64] + b1[lane + 64]);
    } else {
      Xs[r][lane] = (_Float16)0.f;
      Xs[r][lane + 32] = (_Float16)0.f;
      Xs[r][lane + 64] = (_Float16)0.f;
    }
  }

  for (int half = 0; half < 2; ++half) {
    __syncthreads();
    for (int idx = tid; idx < 144 * 96; idx += 256) {   // WT[n][k] = qkv_w[k][...]
      int n = idx / 96, k = idx % 96;
      WT[n][k] = (_Float16)qkv_w[k * 288 + half * 144 + n];
    }
    __syncthreads();
    for (int t = wave; t < 36; t += 8) {
      int mt = t & 3, nt = t >> 2;
      v8f acc = {};
#pragma unroll
      for (int kt = 0; kt < 3; ++kt)
        acc = wmma16(frag_a(&Xs[0][0], 96, mt * 16, kt * 32),
                     frag_bt(&WT[0][0], 96, kt * 32, nt * 16), acc);
      int n_g = half * 144 + nt * 16 + (lane & 15);
      int m3 = n_g / 96, hd_ = (n_g % 96) / 32, dch = n_g & 31;
      float bias = qkv_b[n_g];
      float scale = (m3 == 0) ? 0.1767766952966369f : 1.f;  // HD^-0.5 on q
      size_t base = ((((size_t)w * NH + hd_) * 3 + m3) * Ntok) * HD + dch;
      int rlo = (lane & 16) ? 8 : 0;
#pragma unroll
      for (int g = 0; g < 8; ++g) {
        int row = mt * 16 + g + rlo;
        if (row < Ntok)
          qkvbuf[base + (size_t)row * HD] = (_Float16)((acc[g] + bias) * scale);
      }
    }
  }
}

// ---------------- Kernel 2: windowed attention, one wave per (window, head) ---
__global__ __launch_bounds__(32) void k_attn(const _Float16* __restrict__ qkvbuf,
                                             const float* __restrict__ rpb,
                                             _Float16* __restrict__ obuf) {
  __shared__ __align__(32) _Float16 Qs[64][32];
  __shared__ __align__(32) _Float16 Ks[64][32];
  __shared__ __align__(32) _Float16 VT[32][64];
  __shared__ float Ss[64][64];
  __shared__ __align__(32) _Float16 Ps[64][64];

  int wh_id = blockIdx.x;
  int w = wh_id / NH, head = wh_id % NH;
  int widx = w & 63, wwh = widx >> 3, www = widx & 7;
  int lane = threadIdx.x & 31;
  int rlo = (lane & 16) ? 8 : 0;

  const _Float16* qp = qkvbuf + (((size_t)w * NH + head) * 3 + 0) * Ntok * HD;
  const _Float16* kp = qp + Ntok * HD;
  const _Float16* vp = kp + Ntok * HD;

  // Q,K tiles (49*32 f16, contiguous) via TDM; overlap with V transpose gather.
  tdm_load_2d_f16((unsigned)(size_t)&Qs[0][0], qp, Ntok * HD, 1, Ntok * HD);
  tdm_load_2d_f16((unsigned)(size_t)&Ks[0][0], kp, Ntok * HD, 1, Ntok * HD);
  for (int idx = lane; idx < 64 * 32; idx += 32) {
    int r = idx >> 5, d = idx & 31;
    if (r >= Ntok) { Qs[r][d] = (_Float16)0.f; Ks[r][d] = (_Float16)0.f; }
    VT[d][r] = (r < Ntok) ? vp[r * HD + d] : (_Float16)0.f;
  }
  __builtin_amdgcn_s_wait_tensorcnt(0);
  __syncthreads();

  // S = Q*K^T (+ rel-pos bias + shift mask; -1e4 on pad columns)
  for (int t = 0; t < 16; ++t) {
    int mt = t >> 2, nt = t & 3;
    v8f acc = {};
    acc = wmma16(frag_a(&Qs[0][0], 32, mt * 16, 0),
                 frag_bt(&Ks[0][0], 32, 0, nt * 16), acc);
    int m = nt * 16 + (lane & 15);
#pragma unroll
    for (int g = 0; g < 8; ++g) {
      int n = mt * 16 + g + rlo;
      float val;
      if (n < Ntok && m < Ntok) {
        int ridx = ((n / 7 - m / 7) + 6) * 13 + ((n % 7 - m % 7) + 6);
        float bias = rpb[ridx * NH + head];
        int hn = wwh * 7 + n / 7, wn = www * 7 + n % 7;
        int hm = wwh * 7 + m / 7, wm = www * 7 + m % 7;
        int idn = ((hn < 49) ? 0 : (hn < 53) ? 1 : 2) * 3 +
                  ((wn < 49) ? 0 : (wn < 53) ? 1 : 2);
        int idm = ((hm < 49) ? 0 : (hm < 53) ? 1 : 2) * 3 +
                  ((wm < 49) ? 0 : (wm < 53) ? 1 : 2);
        val = acc[g] + bias + ((idn != idm) ? -100.f : 0.f);
      } else {
        val = -10000.f;
      }
      Ss[n][m] = val;
    }
  }
  __syncthreads();

  for (int r = 0; r < 64; ++r) {   // softmax rows (pad rows uniform, discarded)
    float v0 = Ss[r][lane], v1 = Ss[r][lane + 32];
    float mx = wave_max(fmaxf(v0, v1));
    float e0 = __expf(v0 - mx), e1 = __expf(v1 - mx);
    float inv = 1.f / wave_sum(e0 + e1);
    Ps[r][lane]      = (_Float16)(e0 * inv);
    Ps[r][lane + 32] = (_Float16)(e1 * inv);
  }
  __syncthreads();

  // O = P * V
  for (int t = 0; t < 8; ++t) {
    int mt = t >> 1, nt = t & 1;
    v8f acc = {};
#pragma unroll
    for (int kt = 0; kt < 2; ++kt)
      acc = wmma16(frag_a(&Ps[0][0], 64, mt * 16, kt * 32),
                   frag_bt(&VT[0][0], 64, kt * 32, nt * 16), acc);
    int d = nt * 16 + (lane & 15);
#pragma unroll
    for (int g = 0; g < 8; ++g) {
      int row = mt * 16 + g + rlo;
      if (row < Ntok)
        obuf[((size_t)w * Ntok + row) * Cc + head * HD + d] = (_Float16)acc[g];
    }
  }
}

// ---------------- Kernel 3: proj + window-reverse/unshift + residual ----------
__global__ __launch_bounds__(128) void k_proj(const _Float16* __restrict__ obuf,
                                              const float* __restrict__ proj_w,
                                              const float* __restrict__ proj_b,
                                              const float* __restrict__ x,
                                              float* __restrict__ out) {
  __shared__ __align__(32) _Float16 Os[64][96];
  __shared__ __align__(32) _Float16 WT[96][96];
  int w = blockIdx.x;
  int b = w >> 6, widx = w & 63, wh = widx >> 3, ww2 = widx & 7;
  int tid = threadIdx.x, wave = tid >> 5, lane = tid & 31;

  // Attention-output block (49x96 f16, contiguous) via TDM (1D tile).
  if (wave == 0)
    tdm_load_2d_f16((unsigned)(size_t)&Os[0][0], obuf + (size_t)w * Ntok * Cc,
                    Ntok * Cc, 1, Ntok * Cc);
  for (int idx = tid; idx < 15 * 96; idx += 128) {  // zero pad rows 49..63
    int r = Ntok + idx / 96, c = idx % 96;
    Os[r][c] = (_Float16)0.f;
  }
  for (int idx = tid; idx < 96 * 96; idx += 128) {
    int n = idx / 96, k = idx % 96;
    WT[n][k] = (_Float16)proj_w[k * 96 + n];
  }
  if (wave == 0) __builtin_amdgcn_s_wait_tensorcnt(0);
  __syncthreads();

  for (int t = wave; t < 24; t += 4) {
    int mt = t & 3, nt = t >> 2;
    v8f acc = {};
#pragma unroll
    for (int kt = 0; kt < 3; ++kt)
      acc = wmma16(frag_a(&Os[0][0], 96, mt * 16, kt * 32),
                   frag_bt(&WT[0][0], 96, kt * 32, nt * 16), acc);
    int c = nt * 16 + (lane & 15);
    float bias = proj_b[c];
    int rlo = (lane & 16) ? 8 : 0;
#pragma unroll
    for (int g = 0; g < 8; ++g) {
      int row = mt * 16 + g + rlo;
      if (row < Ntok) {
        int th = row / 7, tw = row % 7;
        int oh = (wh * 7 + th + SH3) % Hh;
        int ow = (ww2 * 7 + tw + SH3) % Wd;
        size_t idx = ((size_t)b * (Hh * Wd) + oh * Wd + ow) * Cc + c;
        out[idx] = x[idx] + acc[g] + bias;
      }
    }
  }
}

// ---------------- Kernel 4a: LN2 + FC1 + GELU -> hidden (f16) ----------------
__global__ __launch_bounds__(256) void k_mlp1(const float* __restrict__ xp,
                                              const float* __restrict__ g2,
                                              const float* __restrict__ b2,
                                              const float* __restrict__ fc1_w,
                                              const float* __restrict__ fc1_b,
                                              _Float16* __restrict__ hid) {
  __shared__ __align__(32) _Float16 Xs[64][96];
  __shared__ __align__(32) _Float16 WT[192][96];
  int tid = threadIdx.x, wave = tid >> 5, lane = tid & 31;
  size_t row0 = (size_t)blockIdx.x * 64;
#pragma unroll
  for (int i = 0; i < 8; ++i) {
    int r = wave * 8 + i;
    const float* px = xp + (row0 + r) * Cc;
    float v0 = px[lane], v1 = px[lane + 32], v2 = px[lane + 64];
    float mu = wave_sum(v0 + v1 + v2) * (1.f / 96.f);
    float d0 = v0 - mu, d1 = v1 - mu, d2 = v2 - mu;
    float var = wave_sum(d0 * d0 + d1 * d1 + d2 * d2) * (1.f / 96.f);
    float rs = rsqrtf(var + 1e-5f);
    Xs[r][lane]      = (_Float16)(d0 * rs * g2[lane]      + b2[lane]);
    Xs[r][lane + 32] = (_Float16)(d1 * rs * g2[lane + 32] + b2[lane + 32]);
    Xs[r][lane + 64] = (_Float16)(d2 * rs * g2[lane + 64] + b2[lane + 64]);
  }
  for (int half = 0; half < 2; ++half) {
    __syncthreads();
    for (int idx = tid; idx < 192 * 96; idx += 256) {
      int n = idx / 96, k = idx % 96;
      WT[n][k] = (_Float16)fc1_w[k * MLPH + half * 192 + n];
    }
    __syncthreads();
    for (int t = wave; t < 48; t += 8) {
      int mt = t & 3, nt = t >> 2;
      v8f acc = {};
#pragma unroll
      for (int kt = 0; kt < 3; ++kt)
        acc = wmma16(frag_a(&Xs[0][0], 96, mt * 16, kt * 32),
                     frag_bt(&WT[0][0], 96, kt * 32, nt * 16), acc);
      int n_g = half * 192 + nt * 16 + (lane & 15);
      float bias = fc1_b[n_g];
      int rlo = (lane & 16) ? 8 : 0;
#pragma unroll
      for (int g = 0; g < 8; ++g) {
        int row = mt * 16 + g + rlo;
        float v = acc[g] + bias;
        v = 0.5f * v * (1.f + erff(v * 0.70710678118654752f));  // exact GELU
        hid[(row0 + row) * MLPH + n_g] = (_Float16)v;
      }
    }
  }
}

// ------- Kernel 4b: FC2 + residual; TDM double-buffered hidden chunks --------
__global__ __launch_bounds__(256) void k_mlp2(const _Float16* __restrict__ hid,
                                              const float* __restrict__ fc2_w,
                                              const float* __restrict__ fc2_b,
                                              float* __restrict__ out) {
  __shared__ __align__(32) _Float16 Hs[2][64][96];
  __shared__ __align__(32) _Float16 WT[96][96];
  int tid = threadIdx.x, wave = tid >> 5, lane = tid & 31;
  size_t row0 = (size_t)blockIdx.x * 64;
  v8f acc[3] = {};

  // Prime the pipeline: DMA chunk 0 while we start staging WT.
  if (wave == 0)
    tdm_load_2d_f16((unsigned)(size_t)&Hs[0][0][0], hid + row0 * MLPH,
                    96, 64, MLPH);

  for (int kc = 0; kc < 4; ++kc) {  // K = 384 in 4 chunks of 96
    __syncthreads();                // prev compute done: WT + next Hs buf free
    for (int idx = tid; idx < 96 * 96; idx += 256) {
      int n = idx / 96, k = idx % 96;
      WT[n][k] = (_Float16)fc2_w[(kc * 96 + k) * Cc + n];
    }
    if (wave == 0) {
      if (kc < 3)   // DMA chunk kc+1 into the other buffer (overlaps compute)
        tdm_load_2d_f16((unsigned)(size_t)&Hs[(kc + 1) & 1][0][0],
                        hid + row0 * MLPH + (kc + 1) * 96, 96, 64, MLPH);
      if (kc < 3) __builtin_amdgcn_s_wait_tensorcnt(1);  // chunk kc landed
      else        __builtin_amdgcn_s_wait_tensorcnt(0);
    }
    __syncthreads();
#pragma unroll
    for (int i = 0; i < 3; ++i) {
      int t = wave + 8 * i;  // 24 tiles / 8 waves
      int mt = t & 3, nt = t >> 2;
#pragma unroll
      for (int kt = 0; kt < 3; ++kt)
        acc[i] = wmma16(frag_a(&Hs[kc & 1][0][0], 96, mt * 16, kt * 32),
                        frag_bt(&WT[0][0], 96, kt * 32, nt * 16), acc[i]);
    }
  }
  int rlo = (lane & 16) ? 8 : 0;
#pragma unroll
  for (int i = 0; i < 3; ++i) {
    int t = wave + 8 * i;
    int mt = t & 3, nt = t >> 2;
    int c = nt * 16 + (lane & 15);
    float bias = fc2_b[c];
#pragma unroll
    for (int g = 0; g < 8; ++g) {
      int row = mt * 16 + g + rlo;
      size_t idx = (row0 + row) * Cc + c;
      out[idx] = out[idx] + acc[i][g] + bias;  // element-local RMW: safe
    }
  }
}

// ---------------- Launch ----------------
extern "C" void kernel_launch(void* const* d_in, const int* in_sizes, int n_in,
                              void* d_out, int out_size, void* d_ws, size_t ws_size,
                              hipStream_t stream) {
  (void)in_sizes; (void)n_in; (void)out_size; (void)ws_size;
  const float* x      = (const float*)d_in[0];
  const float* n1g    = (const float*)d_in[1];
  const float* n1b    = (const float*)d_in[2];
  const float* qkv_w  = (const float*)d_in[3];
  const float* qkv_b  = (const float*)d_in[4];
  const float* proj_w = (const float*)d_in[5];
  const float* proj_b = (const float*)d_in[6];
  const float* rpb    = (const float*)d_in[7];
  const float* n2g    = (const float*)d_in[8];
  const float* n2b    = (const float*)d_in[9];
  const float* fc1_w  = (const float*)d_in[10];
  const float* fc1_b  = (const float*)d_in[11];
  const float* fc2_w  = (const float*)d_in[12];
  const float* fc2_b  = (const float*)d_in[13];
  float* out = (float*)d_out;

  _Float16* qkvbuf = (_Float16*)d_ws;                                   // 115.6 MB
  _Float16* obuf   = qkvbuf + (size_t)NWTOT * NH * 3 * Ntok * HD;       //  38.5 MB
  _Float16* hid    = obuf + (size_t)NWTOT * Ntok * Cc;                  // 154.1 MB

  k_ln1_qkv<<<NWTOT, 256, 0, stream>>>(x, n1g, n1b, qkv_w, qkv_b, qkvbuf);
  k_attn<<<NWTOT * NH, 32, 0, stream>>>(qkvbuf, rpb, obuf);
  k_proj<<<NWTOT, 128, 0, stream>>>(obuf, proj_w, proj_b, x, out);
  k_mlp1<<<ROWS / 64, 256, 0, stream>>>(out, n2g, n2b, fc1_w, fc1_b, hid);
  k_mlp2<<<ROWS / 64, 256, 0, stream>>>(hid, fc2_w, fc2_b, out);
}
